// DGCNN_Contrast2_10694468567371
// MI455X (gfx1250) — matrix operand order repaired and checked by reading, not compiled, not run on hardware
//
#include <hip/hip_runtime.h>
#include <stdint.h>

#define NPTS  1024
#define BATCH 8
#define KNN   20
#define FINF  (__builtin_inff())

// ---------------- WMMA types (CDNA5 / gfx1250, wave32) ----------------
typedef __attribute__((ext_vector_type(16))) __bf16 v16bf;
typedef __attribute__((ext_vector_type(8)))  float  v8f;

union Frag {
  v16bf          v;
  unsigned short s[16];
  uint4          q[2];
};

__device__ __forceinline__ unsigned short f2bf(float f) {
  unsigned int u = __float_as_uint(f);
  u += 0x7fffu + ((u >> 16) & 1u);           // round-to-nearest-even
  return (unsigned short)(u >> 16);
}

// float atomic max via int/uint trick (dest pre-initialized to -inf, no NaNs)
__device__ __forceinline__ void atomicMaxF(float* addr, float v) {
  if (v >= 0.0f) atomicMax((int*)addr, __float_as_int(v));
  else           atomicMin((unsigned int*)addr, __float_as_uint(v));
}

// ---------------- small utility kernels ----------------
__global__ void fill_f32(float* __restrict__ p, float v, int n) {
  int i = blockIdx.x * blockDim.x + threadIdx.x;
  if (i < n) p[i] = v;
}

__global__ void convert_w(const float* __restrict__ W, unsigned short* __restrict__ Wbf,
                          int O, int C, int Cpad) {
  int i = blockIdx.x * blockDim.x + threadIdx.x;
  if (i >= O * Cpad) return;
  int o = i / Cpad, c = i - o * Cpad;
  Wbf[i] = (c < C) ? f2bf(W[o * C + c]) : (unsigned short)0;
}

__global__ void finalize_stats(const float* __restrict__ st, const float* __restrict__ gamma,
                               const float* __restrict__ beta, float* __restrict__ ss,
                               int O, float invM) {
  int o = blockIdx.x * blockDim.x + threadIdx.x;
  if (o >= O) return;
  float mean = st[o] * invM;
  float var  = st[O + o] * invM - mean * mean;
  float s    = gamma[o] * rsqrtf(var + 1e-5f);
  ss[o]      = s;
  ss[O + o]  = beta[o] - mean * s;
}

// ---------------- operand builders: write bf16 [column][channel(Cpad)] ----------------
// edge features: col = (b, n, k) flat; c<Ch: nbr-center, c<2Ch: center, else 0
__global__ void build_edge_bf16(const float* __restrict__ X, int ib, int is, int Ch,
                                const int* __restrict__ idx, unsigned short* __restrict__ out,
                                int Cpad, int ncol) {
  int i = blockIdx.x * blockDim.x + threadIdx.x;
  if (i >= ncol * Cpad) return;
  int col = i / Cpad, c = i - col * Cpad;
  int b = col / (NPTS * KNN);
  int rem = col - b * (NPTS * KNN);
  int n = rem / KNN;
  int kk = rem - n * KNN;
  float v;
  if (c < Ch) {
    int nbr = idx[(b * NPTS + n) * KNN + kk];
    const float* row = X + (size_t)(b * is + ib + c) * NPTS;
    v = row[nbr] - row[n];
  } else if (c < 2 * Ch) {
    v = X[(size_t)(b * is + ib + (c - Ch)) * NPTS + n];
  } else v = 0.0f;
  out[i] = f2bf(v);
}

// plain transpose: col = (b, n); value = X[b, c, n]
__global__ void build_plain_bf16(const float* __restrict__ X, int ib, int is, int Cin,
                                 unsigned short* __restrict__ out, int Cpad, int ncol) {
  int i = blockIdx.x * blockDim.x + threadIdx.x;
  if (i >= ncol * Cpad) return;
  int col = i / Cpad, c = i - col * Cpad;
  int b = col >> 10, n = col & (NPTS - 1);
  float v = (c < Cin) ? X[(size_t)(b * is + ib + c) * NPTS + n] : 0.0f;
  out[i] = f2bf(v);
}

// concat [glob(2048) | pointfeat(64)] per point, Cpad = 2112
__global__ void build_cat_bf16(const float* __restrict__ glob, const float* __restrict__ pf,
                               int pf_base, int pf_stride, unsigned short* __restrict__ out,
                               int ncol) {
  int i = blockIdx.x * blockDim.x + threadIdx.x;
  if (i >= ncol * 2112) return;
  int col = i / 2112, c = i - col * 2112;
  int b = col >> 10, n = col & (NPTS - 1);
  float v;
  if (c < 2048) v = glob[b * 2048 + c];
  else          v = pf[(size_t)(b * pf_stride + pf_base + (c - 2048)) * NPTS + n];
  out[i] = f2bf(v);
}

// ---------------- kNN: one wave per (b,n), 32 candidate points per lane ----------------
__global__ __launch_bounds__(128) void knn_kernel(const float* __restrict__ X, int base, int stride,
                                                  int C, int* __restrict__ idxout) {
  int wid = blockIdx.x * (blockDim.x >> 5) + (threadIdx.x >> 5);
  if (wid >= BATCH * NPTS) return;
  int lane = threadIdx.x & 31;
  int b = wid >> 10, n = wid & (NPTS - 1);

  float d[32];
#pragma unroll
  for (int ch = 0; ch < 32; ++ch) d[ch] = 0.0f;

  for (int c = 0; c < C; ++c) {
    const float* row = X + (size_t)(b * stride + base + c) * NPTS;
    float xn = row[n];
#pragma unroll
    for (int ch = 0; ch < 32; ++ch) {
      float diff = row[(ch << 5) + lane] - xn;
      d[ch] = fmaf(diff, diff, d[ch]);
    }
  }

  for (int kk = 0; kk < KNN; ++kk) {
    float bv = FINF; int bi = 0x7fffffff;
#pragma unroll
    for (int ch = 0; ch < 32; ++ch) {
      if (d[ch] < bv) { bv = d[ch]; bi = (ch << 5) + lane; }
    }
#pragma unroll
    for (int off = 16; off > 0; off >>= 1) {     // argmin, ties -> lower index
      float ov = __shfl_xor(bv, off, 32);
      int   oi = __shfl_xor(bi, off, 32);
      if (ov < bv || (ov == bv && oi < bi)) { bv = ov; bi = oi; }
    }
    if (lane == 0) idxout[wid * KNN + kk] = bi;
#pragma unroll
    for (int ch = 0; ch < 32; ++ch)
      if (((ch << 5) + lane) == bi) d[ch] = FINF;
  }
}

// ---------------- global max+avg pool over points: one wave per (b,o) ----------------
__global__ __launch_bounds__(128) void glob_reduce(const float* __restrict__ h5, float* __restrict__ gl) {
  int wid = blockIdx.x * (blockDim.x >> 5) + (threadIdx.x >> 5);
  if (wid >= BATCH * 1024) return;
  int lane = threadIdx.x & 31;
  const float* row = h5 + (size_t)wid * NPTS;
  float mx = -FINF, sm = 0.0f;
  for (int i = lane; i < NPTS; i += 32) {
    float v = row[i];
    mx = fmaxf(mx, v);
    sm += v;
  }
#pragma unroll
  for (int off = 16; off > 0; off >>= 1) {
    mx = fmaxf(mx, __shfl_xor(mx, off, 32));
    sm += __shfl_xor(sm, off, 32);
  }
  if (lane == 0) {
    int b = wid >> 10, o = wid & 1023;
    gl[b * 2048 + o]        = mx;
    gl[b * 2048 + 1024 + o] = sm * (1.0f / NPTS);
  }
}

// ------------ GEMM + BN: 32x64 tile per wave (2 row-tiles x 4 col-tiles) -------------
// per K-step: 4 A-loads + 8 B-loads (b128) feed 8 WMMAs; each B reused by both A tiles
enum { PHASE_STATS = 0, PHASE_FINAL = 1 };
enum { ACT_NONE = 0, ACT_RELU = 1, ACT_LRELU = 2 };

template <int PHASE, int ACT, bool MAXK, bool EDGECOLS>
__global__ __launch_bounds__(128) void gemm_bn(
    const unsigned short* __restrict__ Abf,   // weights  [O][Cpad] bf16
    const unsigned short* __restrict__ Bbf,   // operand  [ncol][Cpad] bf16
    int O, int Cpad,
    float* __restrict__ stats,
    const float* __restrict__ ss,
    float* __restrict__ Xout, int out_base, int out_stride,
    int ncol) {
  const int lane = threadIdx.x & 31;
  const int wid  = blockIdx.x * (blockDim.x >> 5) + (threadIdx.x >> 5);
  const int rowGroups = O >> 5;                 // 2 row-tiles (32 out-channels) per wave
  const int colGroups = ncol >> 6;              // 4 col-tiles (64 cols) per wave
  if (wid >= rowGroups * colGroups) return;     // wave-uniform: EXEC all-ones at WMMA
  const int rt   = wid % rowGroups;
  const int cg   = wid / rowGroups;
  const int o0   = rt << 5;
  const int col0 = cg << 6;
  const int half = lane >> 4;
  const int l16  = lane & 15;

  // per-lane fragment base pointers (ISA 16-bit A/B layouts)
  const unsigned short* aPtr0 = Abf + (size_t)(o0 + l16) * Cpad + (half << 3);
  const unsigned short* aPtr1 = aPtr0 + ((size_t)Cpad << 4);          // rows o0+16..o0+31
  const unsigned short* bPtr  = Bbf + (size_t)(col0 + l16) * Cpad + (half << 3);
  const size_t tstr = (size_t)Cpad << 4;        // 16 columns per tile

  v8f acc[2][4] = {};
  for (int c0 = 0; c0 < Cpad; c0 += 32) {
    Frag A0, A1;
    A0.q[0] = *(const uint4*)(aPtr0 + c0);
    A0.q[1] = *(const uint4*)(aPtr0 + c0 + 16);
    A1.q[0] = *(const uint4*)(aPtr1 + c0);
    A1.q[1] = *(const uint4*)(aPtr1 + c0 + 16);
    if (c0 + 32 < Cpad) {                       // stream-prefetch next B K-slice
      __builtin_prefetch(bPtr + c0 + 32, 0, 1);
      __builtin_prefetch(bPtr + tstr + c0 + 32, 0, 1);
      __builtin_prefetch(bPtr + 2 * tstr + c0 + 32, 0, 1);
      __builtin_prefetch(bPtr + 3 * tstr + c0 + 32, 0, 1);
    }
#pragma unroll
    for (int q = 0; q < 4; ++q) {
      Frag B;
      const unsigned short* p = bPtr + q * tstr + c0;
      B.q[0] = *(const uint4*)(p);
      B.q[1] = *(const uint4*)(p + 16);
      acc[0][q] = __builtin_amdgcn_wmma_f32_16x16x32_bf16(false, A0.v, false, B.v, (short)0,
                                                          acc[0][q], false, false);
      acc[1][q] = __builtin_amdgcn_wmma_f32_16x16x32_bf16(false, A1.v, false, B.v, (short)0,
                                                          acc[1][q], false, false);
    }
  }

  if (PHASE == PHASE_STATS) {
#pragma unroll
    for (int rh = 0; rh < 2; ++rh) {
#pragma unroll
      for (int r = 0; r < 8; ++r) {             // D: VGPR r -> channel o0+16*rh + r + 8*half
        float s  = acc[rh][0][r] + acc[rh][1][r] + acc[rh][2][r] + acc[rh][3][r];
        float s2 = acc[rh][0][r] * acc[rh][0][r] + acc[rh][1][r] * acc[rh][1][r] +
                   acc[rh][2][r] * acc[rh][2][r] + acc[rh][3][r] * acc[rh][3][r];
#pragma unroll
        for (int off = 1; off < 16; off <<= 1) {  // reduce 16 columns within each half
          s  += __shfl_xor(s, off, 32);
          s2 += __shfl_xor(s2, off, 32);
        }
        if (l16 == 0) {
          const int o = o0 + (rh << 4) + r + (half << 3);
          atomicAdd(&stats[o], s);
          atomicAdd(&stats[O + o], s2);
        }
      }
    }
  } else {
#pragma unroll
    for (int q = 0; q < 4; ++q) {
      const int colq = col0 + (q << 4) + l16;
      int b, n;
      if (EDGECOLS) {
        b = colq / (NPTS * KNN);
        int rem = colq - b * (NPTS * KNN);
        n = rem / KNN;
      } else {
        b = colq >> 10;
        n = colq & (NPTS - 1);
      }
#pragma unroll
      for (int rh = 0; rh < 2; ++rh) {
#pragma unroll
        for (int r = 0; r < 8; ++r) {
          const int o = o0 + (rh << 4) + r + (half << 3);
          float v = acc[rh][q][r] * ss[o] + ss[O + o];   // fused BN
          if (ACT == ACT_RELU)  v = fmaxf(v, 0.0f);
          if (ACT == ACT_LRELU) v = (v > 0.0f) ? v : 0.2f * v;
          float* dst = Xout + (size_t)(b * out_stride + out_base + o) * NPTS + n;
          if (MAXK) atomicMaxF(dst, v);                  // max over the 20 neighbors
          else      *dst = v;
        }
      }
    }
  }
}

// ---------------- host orchestration ----------------
extern "C" void kernel_launch(void* const* d_in, const int* in_sizes, int n_in,
                              void* d_out, int out_size, void* d_ws, size_t ws_size,
                              hipStream_t stream) {
  (void)in_sizes; (void)n_in; (void)out_size; (void)ws_size;
  const float* x = (const float*)d_in[0];
  const float* W[8]; const float* g[8]; const float* be[8];
  for (int i = 0; i < 8; ++i) {
    W[i]  = (const float*)d_in[1 + i];
    g[i]  = (const float*)d_in[9 + 2 * i];
    be[i] = (const float*)d_in[10 + 2 * i];
  }

  static const int Wo[8]  = {64, 64, 128, 256, 1024, 512, 256, 128};
  static const int Wc[8]  = {6, 128, 128, 256, 512, 2112, 512, 256};
  static const int Wcp[8] = {32, 128, 128, 256, 512, 2112, 512, 256};  // pad C to mult of 32

  char* base = (char*)d_ws;
  size_t off = 0;
  auto alloc = [&](size_t bytes) -> char* {
    char* p = base + off;
    off += (bytes + 255) & ~(size_t)255;
    return p;
  };
  unsigned short* wb[8];
  for (int i = 0; i < 8; ++i) wb[i] = (unsigned short*)alloc((size_t)Wo[i] * Wcp[i] * 2);
  int*   idxb = (int*)  alloc((size_t)BATCH * NPTS * KNN * 4);
  float* hcat = (float*)alloc((size_t)BATCH * 512  * NPTS * 4);  // x1|x2|x3|x4
  float* h5   = (float*)alloc((size_t)BATCH * 1024 * NPTS * 4);
  float* gl   = (float*)alloc((size_t)BATCH * 2048 * 4);
  float* h6   = (float*)alloc((size_t)BATCH * 512  * NPTS * 4);
  float* h7   = (float*)alloc((size_t)BATCH * 256  * NPTS * 4);
  float* st   = (float*)alloc(2048 * 4);
  float* ss   = (float*)alloc(2048 * 4);
  unsigned short* bops = (unsigned short*)alloc((size_t)BATCH * NPTS * KNN * 256 * 2); // 84MB, reused
  float* outp = (float*)d_out;

  for (int i = 0; i < 8; ++i) {
    int ne = Wo[i] * Wcp[i];
    convert_w<<<(ne + 255) / 256, 256, 0, stream>>>(W[i], wb[i], Wo[i], Wc[i], Wcp[i]);
  }

  const int edgeCols  = BATCH * NPTS * KNN;      // 163840 (mult of 64)
  const int plainCols = BATCH * NPTS;            // 8192   (mult of 64)
  const float invMedge  = 1.0f / (float)edgeCols;
  const float invMplain = 1.0f / (float)plainCols;

  { int ne = BATCH * 512 * NPTS;                 // -inf for neighbor-max accumulation
    fill_f32<<<(ne + 255) / 256, 256, 0, stream>>>(hcat, -FINF, ne); }

  // ---- four edge-conv layers
  struct ECfg { const float* xin; int ib, is, Ch, wi, ob; };
  const ECfg L[4] = {
    { x,    0,   3,   3,   0, 0   },
    { hcat, 0,   512, 64,  1, 64  },
    { hcat, 64,  512, 64,  2, 128 },
    { hcat, 128, 512, 128, 3, 256 },
  };
  for (int i = 0; i < 4; ++i) {
    const ECfg c = L[i];
    const int O = Wo[c.wi], Cp = Wcp[c.wi];
    knn_kernel<<<(BATCH * NPTS + 3) / 4, 128, 0, stream>>>(c.xin, c.ib, c.is, c.Ch, idxb);
    { int ne = edgeCols * Cp;
      build_edge_bf16<<<(ne + 255) / 256, 256, 0, stream>>>(c.xin, c.ib, c.is, c.Ch,
                                                            idxb, bops, Cp, edgeCols); }
    fill_f32<<<(2 * O + 255) / 256, 256, 0, stream>>>(st, 0.0f, 2 * O);
    const int waves  = (O >> 5) * (edgeCols >> 6);
    const int blocks = (waves + 3) / 4;
    gemm_bn<PHASE_STATS, ACT_NONE, false, true><<<blocks, 128, 0, stream>>>(
        wb[c.wi], bops, O, Cp, st, nullptr, nullptr, 0, 0, edgeCols);
    finalize_stats<<<(O + 255) / 256, 256, 0, stream>>>(st, g[c.wi], be[c.wi], ss, O, invMedge);
    gemm_bn<PHASE_FINAL, ACT_LRELU, true, true><<<blocks, 128, 0, stream>>>(
        wb[c.wi], bops, O, Cp, nullptr, ss, hcat, c.ob, 512, edgeCols);
  }

  // helper for plain-column layers
  auto mlp = [&](int wi, const unsigned short* bsrc, int Cp,
                 float* dst, int dstride, int actRelu /*0 none,1 relu,2 lrelu*/) {
    const int O = Wo[wi];
    fill_f32<<<(2 * O + 255) / 256, 256, 0, stream>>>(st, 0.0f, 2 * O);
    const int waves  = (O >> 5) * (plainCols >> 6);
    const int blocks = (waves + 3) / 4;
    gemm_bn<PHASE_STATS, ACT_NONE, false, false><<<blocks, 128, 0, stream>>>(
        wb[wi], bsrc, O, Cp, st, nullptr, nullptr, 0, 0, plainCols);
    finalize_stats<<<(O + 255) / 256, 256, 0, stream>>>(st, g[wi], be[wi], ss, O, invMplain);
    if (actRelu == 2)
      gemm_bn<PHASE_FINAL, ACT_LRELU, false, false><<<blocks, 128, 0, stream>>>(
          wb[wi], bsrc, O, Cp, nullptr, ss, dst, 0, dstride, plainCols);
    else if (actRelu == 1)
      gemm_bn<PHASE_FINAL, ACT_RELU, false, false><<<blocks, 128, 0, stream>>>(
          wb[wi], bsrc, O, Cp, nullptr, ss, dst, 0, dstride, plainCols);
    else
      gemm_bn<PHASE_FINAL, ACT_NONE, false, false><<<blocks, 128, 0, stream>>>(
          wb[wi], bsrc, O, Cp, nullptr, ss, dst, 0, dstride, plainCols);
  };

  // ---- layer 5: hcat(512) -> h5(1024), lrelu
  { int ne = plainCols * 512;
    build_plain_bf16<<<(ne + 255) / 256, 256, 0, stream>>>(hcat, 0, 512, 512, bops, 512, plainCols); }
  mlp(4, bops, 512, h5, 1024, 2);

  // ---- global max/avg pooling
  glob_reduce<<<(BATCH * 1024 + 3) / 4, 128, 0, stream>>>(h5, gl);

  // ---- layer 6: [glob(2048) | x2(64)](2112) -> h6(512), relu
  { int ne = plainCols * 2112;
    build_cat_bf16<<<(ne + 255) / 256, 256, 0, stream>>>(gl, hcat, 64, 512, bops, plainCols); }
  mlp(5, bops, 2112, h6, 512, 1);

  // ---- layer 7: h6(512) -> h7(256), relu
  { int ne = plainCols * 512;
    build_plain_bf16<<<(ne + 255) / 256, 256, 0, stream>>>(h6, 0, 512, 512, bops, 512, plainCols); }
  mlp(6, bops, 512, h7, 256, 1);

  // ---- layer 8: h7(256) -> d_out(128), bn only
  { int ne = plainCols * 256;
    build_plain_bf16<<<(ne + 255) / 256, 256, 0, stream>>>(h7, 0, 256, 256, bops, 256, plainCols); }
  mlp(7, bops, 256, outp, 128, 0);
}
